// GineLayer_87393994539132
// MI455X (gfx1250) — compile-verified, matrix operand order checked by reading, not run on hardware
//
#include <hip/hip_runtime.h>

// Problem constants (match the JAX reference exactly).
#define NNODES 50000
#define NEDGES 640000
#define DIM    128     // hidden dim D
#define HID    256     // MLP hidden H = 2*D
#define BN_EPS 1e-5f

typedef __attribute__((ext_vector_type(16))) __bf16        v16bf;
typedef __attribute__((ext_vector_type(8)))  float         v8f;
typedef __attribute__((ext_vector_type(4)))  float         f4;
typedef __attribute__((ext_vector_type(4)))  unsigned int  u4;
typedef __attribute__((ext_vector_type(2)))  unsigned int  u2;

union Frag { v16bf v; u4 q[2]; };   // 2 x b128 loads fill one 16-bit WMMA operand

// float -> bf16, round to nearest even.
__device__ __forceinline__ unsigned short f2bf(float f) {
    unsigned int x = __float_as_uint(f);
    x += 0x7FFFu + ((x >> 16) & 1u);
    return (unsigned short)(x >> 16);
}

// ---------------------------------------------------------------------------
// K0: z = (1 + eps) * h  (fuses GINE's "(1+eps)*x + aggr" init), zero BN stats
// ---------------------------------------------------------------------------
__global__ void k_init(const float* __restrict__ h, const float* __restrict__ eps,
                       float* __restrict__ z, float* __restrict__ stats) {
    const int gid = blockIdx.x * blockDim.x + threadIdx.x;
    if (gid < 768) stats[gid] = 0.0f;          // colsum1/colsq1/colsum2/colsq2
    const float s = 1.0f + eps[0];
    ((f4*)z)[gid] = s * ((const f4*)h)[gid];   // gid < N*D/4 by grid sizing
}

// ---------------------------------------------------------------------------
// K1: msg = relu(h[src] + edge_attr), scatter-add into z[dst].
// One wave32 per edge; lane = one float4 chunk of D=128.
// edge_attr / edge_index are single-use streams -> non-temporal loads so they
// do not evict the L2-resident h/z gather/scatter working set (192MB L2).
// ---------------------------------------------------------------------------
__global__ void k_edge(const float* __restrict__ h, const int* __restrict__ ei,
                       const float* __restrict__ ea, float* __restrict__ z) {
    const int e    = blockIdx.x * 8 + (threadIdx.x >> 5);
    const int lane = threadIdx.x & 31;
    const int src = __builtin_nontemporal_load(ei + e);           // row 0
    const int dst = __builtin_nontemporal_load(ei + NEDGES + e);  // row 1
    const f4 hv = ((const f4*)(h + (size_t)src * DIM))[lane];
    const f4 av = __builtin_nontemporal_load(((const f4*)(ea + (size_t)e * DIM)) + lane);
    f4 m = hv + av;
    m.x = fmaxf(m.x, 0.0f); m.y = fmaxf(m.y, 0.0f);
    m.z = fmaxf(m.z, 0.0f); m.w = fmaxf(m.w, 0.0f);
    float* zp = z + (size_t)dst * DIM + lane * 4;
    atomicAdd(zp + 0, m.x);
    atomicAdd(zp + 1, m.y);
    atomicAdd(zp + 2, m.z);
    atomicAdd(zp + 3, m.w);
}

// ---------------------------------------------------------------------------
// Pack f32 -> bf16 row-major A operand; AFFINE fuses BN-apply + ReLU
// (column = flat_index mod K, K power of two -> mask).
// ---------------------------------------------------------------------------
template <bool AFFINE>
__global__ void k_pack_bf16(const float* __restrict__ src, unsigned short* __restrict__ dst,
                            const float* __restrict__ scale, const float* __restrict__ shift,
                            int kmask) {
    const int i = blockIdx.x * blockDim.x + threadIdx.x;   // one float4 / 4 bf16
    f4 v = ((const f4*)src)[i];
    if (AFFINE) {
        const int c = (i * 4) & kmask;
        v.x = fmaxf(fmaf(v.x, scale[c],     shift[c]),     0.0f);
        v.y = fmaxf(fmaf(v.y, scale[c + 1], shift[c + 1]), 0.0f);
        v.z = fmaxf(fmaf(v.z, scale[c + 2], shift[c + 2]), 0.0f);
        v.w = fmaxf(fmaf(v.w, scale[c + 3], shift[c + 3]), 0.0f);
    }
    u2 o;
    o.x = (unsigned)f2bf(v.x) | ((unsigned)f2bf(v.y) << 16);
    o.y = (unsigned)f2bf(v.z) | ((unsigned)f2bf(v.w) << 16);
    ((u2*)dst)[i] = o;
}

// ---------------------------------------------------------------------------
// Pack weights: B[K,Ncol] f32 row-major -> Bt[Ncol,K] bf16, so the WMMA B
// fragment (lane = column, K-pairs contiguous) is two b128 loads. Tiny.
// ---------------------------------------------------------------------------
__global__ void k_pack_bt(const float* __restrict__ B, unsigned short* __restrict__ Bt,
                          int K, int Ncol) {
    const int tid = blockIdx.x * blockDim.x + threadIdx.x;   // over K*Ncol
    const int n = tid / K;
    const int k = tid - n * K;
    Bt[tid] = f2bf(B[(size_t)k * Ncol + n]);
}

// ---------------------------------------------------------------------------
// GEMM: C[M,Ncol] = Abf[M,K] @ Bt[Ncol,K]^T via v_wmma_f32_16x16x32_bf16.
// One wave computes a 16 x (16*WT) strip: A fragment loaded once per k-step
// (2 x b128), reused across WT column tiles (2 x b128 each) -> WT WMMAs.
// Wave id is uniform -> EXEC all-1s at every WMMA (ISA requirement).
// Epilogue accumulates per-column sum / sumsq for the following BatchNorm.
// ---------------------------------------------------------------------------
template <int WT>
__global__ void k_gemm_bf16(const unsigned short* __restrict__ Abf,
                            const unsigned short* __restrict__ Bt,
                            float* __restrict__ C,
                            float* __restrict__ colsum, float* __restrict__ colsq,
                            int M, int K, int Ncol) {
    const int lane = threadIdx.x & 31;
    const int wid  = (blockIdx.x * blockDim.x + threadIdx.x) >> 5;  // wave-uniform
    const int stripsN = Ncol / (16 * WT);
    if (wid >= (M >> 4) * stripsN) return;

    const int tm = wid / stripsN;
    const int ts = wid - tm * stripsN;
    const int l15   = lane & 15;
    const int khalf = (lane >> 4) ? 8 : 0;     // K-half split at lane 16 (ISA 7.12.2)

    const unsigned short* Ar = Abf + (size_t)(tm * 16 + l15) * K;

    v8f acc[WT] = {};
    for (int k0 = 0; k0 < K; k0 += 32) {
        __builtin_prefetch(Ar + k0 + 32, 0, 1);          // global_prefetch_b8
        Frag a;
        a.q[0] = *(const u4*)(Ar + k0 + khalf);          // K = khalf+0..7   (pairs)
        a.q[1] = *(const u4*)(Ar + k0 + khalf + 16);     // K = khalf+16..23
#pragma unroll
        for (int t = 0; t < WT; ++t) {
            const unsigned short* Bc = Bt + (size_t)((ts * WT + t) * 16 + l15) * K;
            Frag b;
            b.q[0] = *(const u4*)(Bc + k0 + khalf);
            b.q[1] = *(const u4*)(Bc + k0 + khalf + 16);
            acc[t] = __builtin_amdgcn_wmma_f32_16x16x32_bf16(
                /*neg_a=*/false, a.v, /*neg_b=*/false, b.v,
                /*c_mod=*/(short)0, acc[t], /*reuse_a=*/false, /*reuse_b=*/false);
        }
    }

    // C/D layout: VGPR r holds M = tm*16 + r (+8 for lanes 16..31), N = lane&15.
    const int rbase = tm * 16 + ((lane >> 4) ? 8 : 0);
#pragma unroll
    for (int t = 0; t < WT; ++t) {
        const int col = (ts * WT + t) * 16 + l15;
        float s = 0.0f, sq = 0.0f;
#pragma unroll
        for (int r = 0; r < 8; ++r) { const float v = acc[t][r]; s += v; sq += v * v; }
        atomicAdd(&colsum[col], s);
        atomicAdd(&colsq[col], sq);
#pragma unroll
        for (int r = 0; r < 8; ++r)
            C[(size_t)(rbase + r) * Ncol + col] = acc[t][r];
    }
}

// ---------------------------------------------------------------------------
// BN params: scale = gamma * rsqrt(var + eps), shift = beta - mean * scale
// ---------------------------------------------------------------------------
__global__ void k_bnparams(const float* __restrict__ colsum, const float* __restrict__ colsq,
                           const float* __restrict__ gamma, const float* __restrict__ beta,
                           float* __restrict__ scale, float* __restrict__ shift,
                           int ncols, float invN) {
    const int i = blockIdx.x * blockDim.x + threadIdx.x;
    if (i >= ncols) return;
    const float mean = colsum[i] * invN;
    const float var  = fmaxf(colsq[i] * invN - mean * mean, 0.0f);
    const float sc   = gamma[i] * rsqrtf(var + BN_EPS);
    scale[i] = sc;
    shift[i] = beta[i] - mean * sc;
}

// ---------------------------------------------------------------------------
// K6: out = relu(BN2(t2)) + h   (elementwise over N*D)
// ---------------------------------------------------------------------------
__global__ void k_final(const float* __restrict__ t2, const float* __restrict__ scale,
                        const float* __restrict__ shift, const float* __restrict__ h,
                        float* __restrict__ out) {
    const size_t i = (size_t)blockIdx.x * blockDim.x + threadIdx.x;  // < N*D
    const int c = (int)(i & (DIM - 1));
    out[i] = fmaxf(fmaf(t2[i], scale[c], shift[c]), 0.0f) + h[i];
}

// ---------------------------------------------------------------------------
// Launch. Inputs (setup_inputs order):
//   0:h 1:edge_index 2:edge_attr 3:eps 4:W1 5:g1 6:b1 7:W2 8:g2 9:b2
// Workspace overlays (stream-ordered, no hazards):
//   t2 aliases z   (z dead after k_pack_bf16<false>)
//   t1bf aliases zbf region (zbf dead after GEMM1)
// ---------------------------------------------------------------------------
extern "C" void kernel_launch(void* const* d_in, const int* in_sizes, int n_in,
                              void* d_out, int out_size, void* d_ws, size_t ws_size,
                              hipStream_t stream) {
    const float* h   = (const float*)d_in[0];
    const int*   ei  = (const int*)  d_in[1];
    const float* ea  = (const float*)d_in[2];
    const float* eps = (const float*)d_in[3];
    const float* W1  = (const float*)d_in[4];
    const float* g1  = (const float*)d_in[5];
    const float* b1  = (const float*)d_in[6];
    const float* W2  = (const float*)d_in[7];
    const float* g2  = (const float*)d_in[8];
    const float* b2  = (const float*)d_in[9];
    float* out = (float*)d_out;

    char* p = (char*)d_ws;
    float* z  = (float*)p;                   // [N*D] f32; later reused as t2
    float* t2 = z;
    p += (size_t)NNODES * DIM * sizeof(float);                    // 25.6 MB
    float* t1 = (float*)p;                   // [N*H] f32
    p += (size_t)NNODES * HID * sizeof(float);                    // 51.2 MB
    unsigned short* zbf  = (unsigned short*)p;   // [N*D] bf16 (first half)
    unsigned short* t1bf = (unsigned short*)p;   // [N*H] bf16 (overlays zbf)
    p += (size_t)NNODES * HID * sizeof(unsigned short);           // 25.6 MB
    unsigned short* W1t = (unsigned short*)p; p += DIM * HID * sizeof(unsigned short);
    unsigned short* W2t = (unsigned short*)p; p += HID * DIM * sizeof(unsigned short);
    float* stats   = (float*)p;  p += 768 * sizeof(float);
    float* colsum1 = stats;                // [HID]
    float* colsq1  = colsum1 + HID;        // [HID]
    float* colsum2 = colsq1  + HID;        // [DIM]
    float* colsq2  = colsum2 + DIM;        // [DIM]
    float* scale1 = (float*)p; p += HID * sizeof(float);
    float* shift1 = (float*)p; p += HID * sizeof(float);
    float* scale2 = (float*)p; p += DIM * sizeof(float);
    float* shift2 = (float*)p; p += DIM * sizeof(float);

    const float invN = 1.0f / (float)NNODES;

    // Weight packs (independent, tiny).
    k_pack_bt<<<dim3((DIM * HID) / 256), dim3(256), 0, stream>>>(W1, W1t, DIM, HID);
    k_pack_bt<<<dim3((HID * DIM) / 256), dim3(256), 0, stream>>>(W2, W2t, HID, DIM);

    // z = (1+eps)*h ; zero BN stats.
    k_init<<<dim3((NNODES * DIM / 4) / 256), dim3(256), 0, stream>>>(h, eps, z, stats);

    // Edge messages + scatter-add (bandwidth-critical phase, NT streams).
    k_edge<<<dim3(NEDGES / 8), dim3(256), 0, stream>>>(h, ei, ea, z);

    // Pack z -> bf16.
    k_pack_bf16<false><<<dim3((NNODES * DIM / 4) / 256), dim3(256), 0, stream>>>(
        z, zbf, nullptr, nullptr, 0);

    // GEMM1: t1 = z @ W1 (+ BN1 stats).  Waves = 3125 * (256/64) = 12500.
    k_gemm_bf16<4><<<dim3((12500 + 7) / 8), dim3(256), 0, stream>>>(
        zbf, W1t, t1, colsum1, colsq1, NNODES, DIM, HID);

    // BN1 affine params.
    k_bnparams<<<dim3(1), dim3(HID), 0, stream>>>(colsum1, colsq1, g1, b1,
                                                  scale1, shift1, HID, invN);

    // Pack t1 -> bf16 with fused BN1-apply + ReLU.
    k_pack_bf16<true><<<dim3((NNODES * HID / 4) / 256), dim3(256), 0, stream>>>(
        t1, t1bf, scale1, shift1, HID - 1);

    // GEMM2: t2 = relu(BN1(t1)) @ W2 (+ BN2 stats). Waves = 3125 * (128/64) = 6250.
    k_gemm_bf16<4><<<dim3((6250 + 7) / 8), dim3(256), 0, stream>>>(
        t1bf, W2t, t2, colsum2, colsq2, NNODES, HID, DIM);

    // BN2 affine params.
    k_bnparams<<<dim3(1), dim3(DIM), 0, stream>>>(colsum2, colsq2, g2, b2,
                                                  scale2, shift2, DIM, invN);

    // out = relu(BN2(t2)) + h.
    k_final<<<dim3((NNODES * DIM) / 256), dim3(256), 0, stream>>>(t2, scale2, shift2, h, out);
}
